// MixtureOfExperts_45140106281026
// MI455X (gfx1250) — matrix operand order repaired and checked by reading, not compiled, not run on hardware
//
#include <hip/hip_runtime.h>
#include <math.h>

#define N_TOK 8192
#define DIM   1024
#define OUTD  1024
#define NE    8
#define TOPK  2
#define HID   2048
#define GHID  16

typedef unsigned short ushort_t;
typedef __attribute__((ext_vector_type(8)))  __bf16 v8bf;
typedef __attribute__((ext_vector_type(16))) __bf16 v16bf;
typedef __attribute__((ext_vector_type(8)))  float  v8f;

__device__ __forceinline__ ushort_t f2bf(float f) {
  unsigned u = __float_as_uint(f);
  u += 0x7FFFu + ((u >> 16) & 1u);           // round-to-nearest-even
  return (ushort_t)(u >> 16);
}
__device__ __forceinline__ float gelu_exact(float v) {
  return 0.5f * v * (1.0f + erff(v * 0.70710678118654752f));
}
__device__ __forceinline__ v16bf cat8(v8bf lo, v8bf hi) {
  return __builtin_shufflevector(lo, hi, 0,1,2,3,4,5,6,7,8,9,10,11,12,13,14,15);
}

// ---------------- Gate: gelu(x@gw1+b)@gw2+b -> softmax -> top2 -> buckets ---
__global__ __launch_bounds__(256)
void gate_kernel(const float* __restrict__ x,  const float* __restrict__ gw1,
                 const float* __restrict__ gb1, const float* __restrict__ gw2,
                 const float* __restrict__ gb2,
                 float* __restrict__ topw_f, int* __restrict__ counts,
                 int* __restrict__ bucket_aid) {
  const int wave = threadIdx.x >> 5;
  const int lane = threadIdx.x & 31;
  const int n = blockIdx.x * 8 + wave;

  float acc[GHID];
  #pragma unroll
  for (int j = 0; j < GHID; ++j) acc[j] = 0.0f;

  const float* xr = x + (size_t)n * DIM;
  for (int d = lane; d < DIM; d += 32) {
    float xv = xr[d];
    const float* wr = gw1 + (size_t)d * GHID;
    #pragma unroll
    for (int j = 0; j < GHID; ++j) acc[j] += xv * wr[j];
  }
  #pragma unroll
  for (int j = 0; j < GHID; ++j)
    for (int off = 16; off > 0; off >>= 1)
      acc[j] += __shfl_xor(acc[j], off, 32);

  if (lane == 0) {
    float g2[NE];
    #pragma unroll
    for (int e = 0; e < NE; ++e) g2[e] = gb2[e];
    #pragma unroll
    for (int j = 0; j < GHID; ++j) {
      float g = gelu_exact(acc[j] + gb1[j]);
      #pragma unroll
      for (int e = 0; e < NE; ++e) g2[e] += g * gw2[j * NE + e];
    }
    float mx = g2[0];
    #pragma unroll
    for (int e = 1; e < NE; ++e) mx = fmaxf(mx, g2[e]);
    float p[NE];
    #pragma unroll
    for (int e = 0; e < NE; ++e) p[e] = expf(g2[e] - mx);
    int i0 = 0;
    #pragma unroll
    for (int e = 1; e < NE; ++e) if (p[e] > p[i0]) i0 = e;
    int i1 = (i0 == 0) ? 1 : 0;
    #pragma unroll
    for (int e = 0; e < NE; ++e) if (e != i0 && p[e] > p[i1]) i1 = e;
    float w0 = p[i0], w1v = p[i1];
    float inv = 1.0f / (w0 + w1v);
    w0 *= inv; w1v *= inv;
    int aid0 = n * TOPK, aid1 = n * TOPK + 1;
    topw_f[aid0] = w0;  topw_f[aid1] = w1v;
    int s0 = atomicAdd(&counts[i0], 1); bucket_aid[i0 * N_TOK + s0] = aid0;
    int s1 = atomicAdd(&counts[i1], 1); bucket_aid[i1 * N_TOK + s1] = aid1;
  }
}

// ---------------- f32 -> bf16 for x ----------------------------------------
__global__ __launch_bounds__(256)
void cvt_x_kernel(const float* __restrict__ x, ushort_t* __restrict__ xb) {
  size_t i = ((size_t)blockIdx.x * blockDim.x + threadIdx.x) * 4;
  float4 v = *(const float4*)(x + i);
  union { ushort_t s[4]; uint2 u; } pk;
  pk.s[0] = f2bf(v.x); pk.s[1] = f2bf(v.y); pk.s[2] = f2bf(v.z); pk.s[3] = f2bf(v.w);
  *(uint2*)(xb + i) = pk.u;
}

// ---------------- transpose + convert per-expert weight matrices -----------
__global__ __launch_bounds__(256)
void transpose_cvt_kernel(const float* __restrict__ src, ushort_t* __restrict__ dst,
                          int rows, int cols) {
  __shared__ float t[32][33];
  const int z = blockIdx.z;
  const float* s = src + (size_t)z * rows * cols;
  ushort_t* d = dst + (size_t)z * rows * cols;
  const int c0 = blockIdx.x * 32, r0 = blockIdx.y * 32;
  const int tx = threadIdx.x, ty = threadIdx.y;
  #pragma unroll
  for (int i = 0; i < 4; ++i)
    t[ty + i * 8][tx] = s[(size_t)(r0 + ty + i * 8) * cols + c0 + tx];
  __syncthreads();
  #pragma unroll
  for (int i = 0; i < 4; ++i)
    d[(size_t)(c0 + ty + i * 8) * rows + r0 + tx] = f2bf(t[tx][ty + i * 8]);
}

// ======================= routed WMMA GEMM tile =============================
// Workgroup tile: 64 rows (bucket slots) x 128 cols, K-chunks of 64, double
// buffered LDS.  8 waves in a 2x4 grid, each wave owns a 32x32 C tile
// (2x2 WMMA fragments) -> 4 v_wmma per k-step per wave, 2:1 ds:wmma ratio.
// Per-thread A/B source pointers are hoisted out of the k-loop so staging is
// pure (base + k0) global_load_b128 -> ds_store_b128.
#define TM    64
#define TN    128
#define KC    64
#define KPAD  72

// GEMM1: h = gelu(x @ w1 + b1), routed
__global__ __launch_bounds__(256)
void moe_gemm1_kernel(const ushort_t* __restrict__ xb,
                      const ushort_t* __restrict__ w1b,   // [E][H][D] bf16
                      const float* __restrict__ b1,       // [E][H]
                      const int* __restrict__ counts,
                      const int* __restrict__ bucket_aid,
                      ushort_t* __restrict__ h_ws) {      // [N*K][H] bf16
  const int e = blockIdx.z;
  const int cnt = counts[e];
  const int m0 = blockIdx.y * TM;
  if (m0 >= cnt) return;
  const int n0 = blockIdx.x * TN;

  __shared__ __align__(16) ushort_t As[2][TM][KPAD];
  __shared__ __align__(16) ushort_t Bs[2][TN][KPAD];
  __shared__ int rows[TM];

  const int tid = threadIdx.x;
  if (tid < TM) {
    int s = m0 + tid;
    rows[tid] = (s < cnt) ? bucket_aid[e * N_TOK + s] : -1;
  }
  __syncthreads();

  const int r8 = tid >> 3, c8 = (tid & 7) * 8;

  // loop-invariant per-thread sources (hoisted out of the k-loop)
  const int aidA0 = rows[r8];
  const int aidA1 = rows[r8 + 32];
  const ushort_t* aPtr0 =
      xb + (size_t)(aidA0 >= 0 ? (aidA0 >> 1) : 0) * DIM + c8;
  const ushort_t* aPtr1 =
      xb + (size_t)(aidA1 >= 0 ? (aidA1 >> 1) : 0) * DIM + c8;
  const ushort_t* wBr =
      w1b + ((size_t)e * HID + n0 + r8) * DIM + c8;

  auto stage = [&](int k0, int buf) {
    uint4 va0 = {0u, 0u, 0u, 0u}, va1 = {0u, 0u, 0u, 0u};
    if (aidA0 >= 0) va0 = *(const uint4*)(aPtr0 + k0);
    if (aidA1 >= 0) va1 = *(const uint4*)(aPtr1 + k0);
    *(uint4*)(&As[buf][r8][c8]) = va0;
    *(uint4*)(&As[buf][r8 + 32][c8]) = va1;
    #pragma unroll
    for (int i = 0; i < 4; ++i)
      *(uint4*)(&Bs[buf][r8 + i * 32][c8]) =
          *(const uint4*)(wBr + (size_t)i * 32 * DIM + k0);
    if (k0 + 2 * KC < DIM)
      __builtin_prefetch(wBr + k0 + 2 * KC, 0, 0);
  };

  const int wid = tid >> 5, lane = tid & 31;
  const int wm = wid >> 2, wn = wid & 3;
  const int lmod = lane & 15, ldiv = lane >> 4;

  v8f acc[2][2];
  #pragma unroll
  for (int i = 0; i < 2; ++i)
    #pragma unroll
    for (int j = 0; j < 2; ++j)
      acc[i][j] = (v8f){0.f,0.f,0.f,0.f,0.f,0.f,0.f,0.f};

  stage(0, 0);
  __syncthreads();
  const int NK = DIM / KC;
  for (int kc = 0; kc < NK; ++kc) {
    const int buf = kc & 1;
    if (kc + 1 < NK) stage((kc + 1) * KC, buf ^ 1);
    #pragma unroll
    for (int ks = 0; ks < 2; ++ks) {
      const int kk = ks * 32;
      v16bf a[2], b[2];
      #pragma unroll
      for (int ms = 0; ms < 2; ++ms) {
        const ushort_t* pa = &As[buf][wm * 32 + ms * 16 + lmod][kk + ldiv * 8];
        a[ms] = cat8(*(const v8bf*)pa, *(const v8bf*)(pa + 16));
      }
      #pragma unroll
      for (int ns = 0; ns < 2; ++ns) {
        const ushort_t* pb = &Bs[buf][wn * 32 + ns * 16 + lmod][kk + ldiv * 8];
        b[ns] = cat8(*(const v8bf*)pb, *(const v8bf*)(pb + 16));
      }
      #pragma unroll
      for (int ms = 0; ms < 2; ++ms)
        #pragma unroll
        for (int ns = 0; ns < 2; ++ns)
          acc[ms][ns] = __builtin_amdgcn_wmma_f32_16x16x32_bf16(
              false, a[ms], false, b[ns], (short)0, acc[ms][ns], false, false);
    }
    __syncthreads();
  }

  #pragma unroll
  for (int ns = 0; ns < 2; ++ns) {
    const int n = n0 + wn * 32 + ns * 16 + lmod;
    const float bias = b1[e * HID + n];
    #pragma unroll
    for (int ms = 0; ms < 2; ++ms) {
      #pragma unroll
      for (int r = 0; r < 8; ++r) {
        const int m = wm * 32 + ms * 16 + ldiv * 8 + r;
        const int aid = rows[m];
        if (aid >= 0)
          h_ws[(size_t)aid * HID + n] = f2bf(gelu_exact(acc[ms][ns][r] + bias));
      }
    }
  }
}

// GEMM2: out += gate_w * (h @ w2 + b2), routed
__global__ __launch_bounds__(256)
void moe_gemm2_kernel(const ushort_t* __restrict__ h_ws,  // [N*K][H] bf16
                      const ushort_t* __restrict__ w2b,   // [E][O][H] bf16
                      const float* __restrict__ b2,       // [E][O]
                      const int* __restrict__ counts,
                      const int* __restrict__ bucket_aid,
                      const float* __restrict__ topw_f,
                      float* __restrict__ out) {
  const int e = blockIdx.z;
  const int cnt = counts[e];
  const int m0 = blockIdx.y * TM;
  if (m0 >= cnt) return;
  const int n0 = blockIdx.x * TN;

  __shared__ __align__(16) ushort_t As[2][TM][KPAD];
  __shared__ __align__(16) ushort_t Bs[2][TN][KPAD];
  __shared__ int rows[TM];

  const int tid = threadIdx.x;
  if (tid < TM) {
    int s = m0 + tid;
    rows[tid] = (s < cnt) ? bucket_aid[e * N_TOK + s] : -1;
  }
  __syncthreads();

  const int r8 = tid >> 3, c8 = (tid & 7) * 8;

  const int aidA0 = rows[r8];
  const int aidA1 = rows[r8 + 32];
  const ushort_t* aPtr0 =
      h_ws + (size_t)(aidA0 >= 0 ? aidA0 : 0) * HID + c8;
  const ushort_t* aPtr1 =
      h_ws + (size_t)(aidA1 >= 0 ? aidA1 : 0) * HID + c8;
  const ushort_t* wBr =
      w2b + ((size_t)e * OUTD + n0 + r8) * HID + c8;

  auto stage = [&](int k0, int buf) {
    uint4 va0 = {0u, 0u, 0u, 0u}, va1 = {0u, 0u, 0u, 0u};
    if (aidA0 >= 0) va0 = *(const uint4*)(aPtr0 + k0);
    if (aidA1 >= 0) va1 = *(const uint4*)(aPtr1 + k0);
    *(uint4*)(&As[buf][r8][c8]) = va0;
    *(uint4*)(&As[buf][r8 + 32][c8]) = va1;
    #pragma unroll
    for (int i = 0; i < 4; ++i)
      *(uint4*)(&Bs[buf][r8 + i * 32][c8]) =
          *(const uint4*)(wBr + (size_t)i * 32 * HID + k0);
    if (k0 + 2 * KC < HID)
      __builtin_prefetch(wBr + k0 + 2 * KC, 0, 0);
  };

  const int wid = tid >> 5, lane = tid & 31;
  const int wm = wid >> 2, wn = wid & 3;
  const int lmod = lane & 15, ldiv = lane >> 4;

  v8f acc[2][2];
  #pragma unroll
  for (int i = 0; i < 2; ++i)
    #pragma unroll
    for (int j = 0; j < 2; ++j)
      acc[i][j] = (v8f){0.f,0.f,0.f,0.f,0.f,0.f,0.f,0.f};

  stage(0, 0);
  __syncthreads();
  const int NK = HID / KC;
  for (int kc = 0; kc < NK; ++kc) {
    const int buf = kc & 1;
    if (kc + 1 < NK) stage((kc + 1) * KC, buf ^ 1);
    #pragma unroll
    for (int ks = 0; ks < 2; ++ks) {
      const int kk = ks * 32;
      v16bf a[2], b[2];
      #pragma unroll
      for (int ms = 0; ms < 2; ++ms) {
        const ushort_t* pa = &As[buf][wm * 32 + ms * 16 + lmod][kk + ldiv * 8];
        a[ms] = cat8(*(const v8bf*)pa, *(const v8bf*)(pa + 16));
      }
      #pragma unroll
      for (int ns = 0; ns < 2; ++ns) {
        const ushort_t* pb = &Bs[buf][wn * 32 + ns * 16 + lmod][kk + ldiv * 8];
        b[ns] = cat8(*(const v8bf*)pb, *(const v8bf*)(pb + 16));
      }
      #pragma unroll
      for (int ms = 0; ms < 2; ++ms)
        #pragma unroll
        for (int ns = 0; ns < 2; ++ns)
          acc[ms][ns] = __builtin_amdgcn_wmma_f32_16x16x32_bf16(
              false, a[ms], false, b[ns], (short)0, acc[ms][ns], false, false);
    }
    __syncthreads();
  }

  #pragma unroll
  for (int ns = 0; ns < 2; ++ns) {
    const int n = n0 + wn * 32 + ns * 16 + lmod;
    const float bias = b2[e * OUTD + n];
    #pragma unroll
    for (int ms = 0; ms < 2; ++ms) {
      #pragma unroll
      for (int r = 0; r < 8; ++r) {
        const int m = wm * 32 + ms * 16 + ldiv * 8 + r;
        const int aid = rows[m];
        if (aid >= 0) {
          float val = acc[ms][ns][r] + bias;
          float w = topw_f[aid];
          atomicAdd(&out[(size_t)(aid >> 1) * OUTD + n], w * val);
        }
      }
    }
  }
}

// ---------------------------------------------------------------------------
extern "C" void kernel_launch(void* const* d_in, const int* in_sizes, int n_in,
                              void* d_out, int out_size, void* d_ws, size_t ws_size,
                              hipStream_t stream) {
  (void)in_sizes; (void)n_in; (void)ws_size;
  const float* x   = (const float*)d_in[0];
  const float* gw1 = (const float*)d_in[1];
  const float* gb1 = (const float*)d_in[2];
  const float* gw2 = (const float*)d_in[3];
  const float* gb2 = (const float*)d_in[4];
  const float* w1  = (const float*)d_in[5];
  const float* b1  = (const float*)d_in[6];
  const float* w2  = (const float*)d_in[7];
  const float* b2  = (const float*)d_in[8];
  float* out = (float*)d_out;

  char* ws = (char*)d_ws;
  size_t off = 0;
  auto walloc = [&](size_t bytes) -> void* {
    void* p = ws + off;
    off = (off + bytes + 255) & ~(size_t)255;
    return p;
  };
  ushort_t* xb     = (ushort_t*)walloc((size_t)N_TOK * DIM * 2);
  ushort_t* w1b    = (ushort_t*)walloc((size_t)NE * HID * DIM * 2);
  ushort_t* w2b    = (ushort_t*)walloc((size_t)NE * OUTD * HID * 2);
  ushort_t* h_ws   = (ushort_t*)walloc((size_t)N_TOK * TOPK * HID * 2);
  float*    topw   = (float*)walloc((size_t)N_TOK * TOPK * 4);
  int*      counts = (int*)walloc(NE * sizeof(int));
  int*      bucket = (int*)walloc((size_t)NE * N_TOK * sizeof(int));

  hipMemsetAsync(counts, 0, NE * sizeof(int), stream);
  hipMemsetAsync(out, 0, (size_t)out_size * sizeof(float), stream);

  gate_kernel<<<N_TOK / 8, 256, 0, stream>>>(x, gw1, gb1, gw2, gb2, topw, counts, bucket);
  cvt_x_kernel<<<(N_TOK * DIM / 4) / 256, 256, 0, stream>>>(x, xb);
  transpose_cvt_kernel<<<dim3(HID / 32, DIM / 32, NE), dim3(32, 8), 0, stream>>>(w1, w1b, DIM, HID);
  transpose_cvt_kernel<<<dim3(OUTD / 32, HID / 32, NE), dim3(32, 8), 0, stream>>>(w2, w2b, HID, OUTD);

  moe_gemm1_kernel<<<dim3(HID / TN, N_TOK / TM, NE), 256, 0, stream>>>(
      xb, w1b, b1, counts, bucket, h_ws);
  moe_gemm2_kernel<<<dim3(OUTD / TN, N_TOK / TM, NE), 256, 0, stream>>>(
      h_ws, w2b, b2, counts, bucket, topw, out);
}
